// RITS_56160992363035
// MI455X (gfx1250) — compile-verified, matrix operand order checked by reading, not run on hardware
//
#include <hip/hip_runtime.h>
#include <hip/hip_bf16.h>

typedef __attribute__((ext_vector_type(16))) __bf16 v16bf;
typedef __attribute__((ext_vector_type(8)))  float  v8f;

#define Bsz 256
#define Ssz 512
#define Fsz 64
#define Hsz 128
#define ROWS 16
#define NBLK 16
#define NTHREADS 256

// ---------------- LDS layout (bytes), all 16B aligned ----------------
#define OFF_WIH    0                        // bf16 [512][128]  LSTM input weight
#define OFF_WHH    (OFF_WIH + 512*128*2)    // bf16 [512][128]  LSTM hidden weight
#define OFF_BIASG  (OFF_WHH + 512*128*2)    // f32  [512]       bih+bhh
#define OFF_BDH    (OFF_BIASG + 512*4)      // f32  [128]
#define OFF_BDM    (OFF_BDH + 128*4)        // f32  [64]
#define OFF_BTR    (OFF_BDM + 64*4)         // f32  [64]
#define OFF_BFR    (OFF_BTR + 64*4)         // f32  [64]
#define OFF_BWC    (OFF_BFR + 64*4)         // f32  [64]
#define OFF_DWDM   (OFF_BWC + 64*4)         // f32  [64]  diag(Wdm)
#define OFF_HF     (OFF_DWDM + 64*4)        // f32  [16][128]  h state
#define OFF_CF     (OFF_HF + 2048*4)        // f32  [16][128]  c state
#define OFF_HBF    (OFF_CF + 2048*4)        // bf16 [16][128]  h (matmul operand)
#define OFF_TBF    (OFF_HBF + 2048*2)       // bf16 [16][64]   t slice
#define OFF_XHAT   (OFF_TBF + 1024*2)       // f32  [16][64]
#define OFF_ZHAT   (OFF_XHAT + 1024*4)      // f32  [16][64]
#define OFF_BETA   (OFF_ZHAT + 1024*4)      // f32  [16][64]
#define OFF_ACAT   (OFF_BETA + 1024*4)      // bf16 [16][128]  concat operand
#define OFF_XCBF   (OFF_ACAT + 2048*2)      // bf16 [16][64]   x_complement
#define OFF_GATES  (OFF_XCBF + 1024*2)      // bf16 [16][512]  LSTM gates
#define OFF_RED    (OFF_GATES + 8192*2)     // f32  [16]       per-wave loss partials
#define LDS_BYTES  (OFF_RED + 16*4)         // = 323,392 B  (<= 327,680 B per WGP)

static __device__ __forceinline__ v8f wmma_bf16(v16bf a, v16bf b, v8f c) {
    return __builtin_amdgcn_wmma_f32_16x16x32_bf16(false, a, false, b, (short)0, c, false, false);
}

// A/B fragment gather from LDS bf16 matrix (row-major, ldk elems per row).
// ISA 7.12.2 bf16 layout: lane(lo/hi) elem e -> K = k0 + hi*8 + e   (e<8)
//                                          K = k0 + 16 + hi*8 + e-8 (e>=8)
// => two contiguous 16-byte chunks -> ds_load_b128 x2.
static __device__ __forceinline__ v16bf lds_frag(const __bf16* base, int ldk, int row, int k0, int hi) {
    const __bf16* p = base + row * ldk + k0 + hi * 8;
    union { v16bf v; uint4 q[2]; } u;
    u.q[0] = *(const uint4*)(p);
    u.q[1] = *(const uint4*)(p + 16);
    return u.v;
}

// One-time global gather of a B fragment (col n of B == row n of W[N][K]) into registers.
static __device__ __forceinline__ v16bf g_fragB(const float* W, int ldk, int n, int k0, int hi, bool zdiag) {
    v16bf v;
#pragma unroll
    for (int e = 0; e < 16; ++e) {
        int k = (e < 8) ? (k0 + hi * 8 + e) : (k0 + 16 + hi * 8 + (e - 8));
        float val = W[(size_t)n * ldk + k];
        if (zdiag && k == n) val = 0.0f;
        v[e] = (__bf16)val;
    }
    return v;
}

static __device__ __forceinline__ float sigmf(float x) { return 1.0f / (1.0f + __expf(-x)); }

__global__ void __launch_bounds__(NTHREADS)
rits_scan_kernel(const float* __restrict__ X, const float* __restrict__ Mm, const float* __restrict__ T,
                 const float* __restrict__ Wdh, const float* __restrict__ bdh,
                 const float* __restrict__ Wdm, const float* __restrict__ bdm,
                 const float* __restrict__ Wtr, const float* __restrict__ btr,
                 const float* __restrict__ Wfr, const float* __restrict__ bfr,
                 const float* __restrict__ Wwc, const float* __restrict__ bwc,
                 const float* __restrict__ Wih, const float* __restrict__ Whh,
                 const float* __restrict__ bih, const float* __restrict__ bhh,
                 float* __restrict__ out, float* __restrict__ wsNum, float* __restrict__ wsMs)
{
    extern __shared__ char smem[];
    __bf16* sWih   = (__bf16*)(smem + OFF_WIH);
    __bf16* sWhh   = (__bf16*)(smem + OFF_WHH);
    float*  sBiasG = (float*) (smem + OFF_BIASG);
    float*  sBdh   = (float*) (smem + OFF_BDH);
    float*  sBdm   = (float*) (smem + OFF_BDM);
    float*  sBtr   = (float*) (smem + OFF_BTR);
    float*  sBfr   = (float*) (smem + OFF_BFR);
    float*  sBwc   = (float*) (smem + OFF_BWC);
    float*  sDwdm  = (float*) (smem + OFF_DWDM);
    float*  sHf    = (float*) (smem + OFF_HF);
    float*  sCf    = (float*) (smem + OFF_CF);
    __bf16* sHbf   = (__bf16*)(smem + OFF_HBF);
    __bf16* sTbf   = (__bf16*)(smem + OFF_TBF);
    float*  sXhat  = (float*) (smem + OFF_XHAT);
    float*  sZhat  = (float*) (smem + OFF_ZHAT);
    float*  sBeta  = (float*) (smem + OFF_BETA);
    __bf16* sAcat  = (__bf16*)(smem + OFF_ACAT);
    __bf16* sXcbf  = (__bf16*)(smem + OFF_XCBF);
    __bf16* sGates = (__bf16*)(smem + OFF_GATES);
    float*  sRed   = (float*) (smem + OFF_RED);

    const int tid  = threadIdx.x;
    const int wv   = tid >> 5;          // wave id 0..7 (wave32)
    const int lane = tid & 31;
    const int ln   = lane & 15;
    const int hi   = (lane >> 4) & 1;
    const int blk  = blockIdx.x;
    const int b0   = blk * ROWS;

    // ---- one-time: stage big LSTM weights to LDS (bf16), biases, states ----
    for (int i = tid; i < 512 * 128; i += NTHREADS) { sWih[i] = (__bf16)Wih[i]; sWhh[i] = (__bf16)Whh[i]; }
    for (int i = tid; i < 512; i += NTHREADS) sBiasG[i] = bih[i] + bhh[i];
    for (int i = tid; i < 128; i += NTHREADS) sBdh[i] = bdh[i];
    if (tid < 64) {
        sBdm[tid] = bdm[tid]; sBtr[tid] = btr[tid]; sBfr[tid] = bfr[tid]; sBwc[tid] = bwc[tid];
        sDwdm[tid] = Wdm[tid * Fsz + tid];
    }
    for (int i = tid; i < 2048; i += NTHREADS) { sHf[i] = 0.0f; sCf[i] = 0.0f; sHbf[i] = (__bf16)0.0f; }

    // ---- one-time: small weights -> per-wave register B fragments ----
    v16bf wdh[2] = {}; v16bf wtrf[4] = {}; v16bf wfrf[2] = {}; v16bf wwcf[4] = {};
    wdh[0] = g_fragB(Wdh, 64, wv * 16 + ln, 0, hi, false);   // gamma_h: N-tile = wv
    wdh[1] = g_fragB(Wdh, 64, wv * 16 + ln, 32, hi, false);
    if (wv < 4) {
#pragma unroll
        for (int ki = 0; ki < 4; ++ki) wtrf[ki] = g_fragB(Wtr, 128, wv * 16 + ln, ki * 32, hi, false);
        wfrf[0] = g_fragB(Wfr, 64, wv * 16 + ln, 0, hi, true);   // zero-diagonal
        wfrf[1] = g_fragB(Wfr, 64, wv * 16 + ln, 32, hi, true);
    } else {
#pragma unroll
        for (int ki = 0; ki < 4; ++ki) wwcf[ki] = g_fragB(Wwc, 128, (wv - 4) * 16 + ln, ki * 32, hi, false);
    }
    __syncthreads();

    // ================= sequential scan over S steps =================
    for (int s = 0; s < Ssz; ++s) {
        float num_loc = 0.0f, ms_loc = 0.0f;

        // (a) stage t slice as bf16; prefetch next-step inputs into GL2
        for (int i = tid; i < ROWS * Fsz; i += NTHREADS) {
            int row = i >> 6, f = i & 63;
            size_t g = ((size_t)(b0 + row) * Ssz + s) * Fsz + f;
            sTbf[i] = (__bf16)T[g];
            if (s + 1 < Ssz) {
                __builtin_prefetch(&X[g + Fsz], 0, 1);
                __builtin_prefetch(&Mm[g + Fsz], 0, 1);
                __builtin_prefetch(&T[g + Fsz], 0, 1);
            }
        }
        __syncthreads();

        // (b) gamma_h = exp(-relu(t @ Wdh^T + bdh)); h *= gamma_h.  8 waves x 1 N-tile.
        {
            v16bf a0 = lds_frag(sTbf, 64, ln, 0, hi);
            v16bf a1 = lds_frag(sTbf, 64, ln, 32, hi);
            v8f acc = {0.f,0.f,0.f,0.f,0.f,0.f,0.f,0.f};
            acc = wmma_bf16(a0, wdh[0], acc);
            acc = wmma_bf16(a1, wdh[1], acc);
            int n = wv * 16 + ln;
#pragma unroll
            for (int r = 0; r < 8; ++r) {
                int mrow = r + hi * 8;
                float v = acc[r] + sBdh[n];
                v = v > 0.0f ? v : 0.0f;
                float hn = sHf[mrow * Hsz + n] * __expf(-v);
                sHf[mrow * Hsz + n] = hn;
                sHbf[mrow * Hsz + n] = (__bf16)hn;
            }
        }
        __syncthreads();

        // (c) waves 0-3: x_hat = h @ Wtr^T + btr.  waves 4-7: gamma_m -> Acat=[gamma_m, m]
        if (wv < 4) {
            v8f acc = {0.f,0.f,0.f,0.f,0.f,0.f,0.f,0.f};
#pragma unroll
            for (int ki = 0; ki < 4; ++ki)
                acc = wmma_bf16(lds_frag(sHbf, 128, ln, ki * 32, hi), wtrf[ki], acc);
            int n = wv * 16 + ln;
#pragma unroll
            for (int r = 0; r < 8; ++r) sXhat[(r + hi * 8) * Fsz + n] = acc[r] + sBtr[n];
        } else {
            for (int i = tid - 128; i < ROWS * Fsz; i += 128) {
                int row = i >> 6, f = i & 63;
                size_t g = ((size_t)(b0 + row) * Ssz + s) * Fsz + f;
                float v = T[g] * sDwdm[f] + sBdm[f];
                v = v > 0.0f ? v : 0.0f;
                sAcat[row * 128 + f]      = (__bf16)__expf(-v);
                sAcat[row * 128 + 64 + f] = (__bf16)Mm[g];
            }
        }
        __syncthreads();

        // (d) loss term 1, x_complement
        for (int i = tid; i < ROWS * Fsz; i += NTHREADS) {
            int row = i >> 6, f = i & 63;
            size_t g = ((size_t)(b0 + row) * Ssz + s) * Fsz + f;
            float xl = X[g], ml = Mm[g], xh = sXhat[i];
            float d = xl - xh;
            num_loc += d * d * ml;
            ms_loc  += ml;
            sXcbf[i] = (__bf16)(ml * xl + (1.0f - ml) * xh);
        }
        __syncthreads();

        // (e) waves 0-3: z_hat = xc @ Wfr_m^T + bfr.  waves 4-7: beta = Acat @ Wwc^T + bwc
        if (wv < 4) {
            v8f acc = {0.f,0.f,0.f,0.f,0.f,0.f,0.f,0.f};
            acc = wmma_bf16(lds_frag(sXcbf, 64, ln, 0, hi), wfrf[0], acc);
            acc = wmma_bf16(lds_frag(sXcbf, 64, ln, 32, hi), wfrf[1], acc);
            int n = wv * 16 + ln;
#pragma unroll
            for (int r = 0; r < 8; ++r) sZhat[(r + hi * 8) * Fsz + n] = acc[r] + sBfr[n];
        } else {
            v8f acc = {0.f,0.f,0.f,0.f,0.f,0.f,0.f,0.f};
#pragma unroll
            for (int ki = 0; ki < 4; ++ki)
                acc = wmma_bf16(lds_frag(sAcat, 128, ln, ki * 32, hi), wwcf[ki], acc);
            int n = (wv - 4) * 16 + ln;
#pragma unroll
            for (int r = 0; r < 8; ++r) sBeta[(r + hi * 8) * Fsz + n] = acc[r] + sBwc[n];
        }
        __syncthreads();

        // (f) loss 2+3, c_hat, c_complement -> output + Acat=[cc, m]
        for (int i = tid; i < ROWS * Fsz; i += NTHREADS) {
            int row = i >> 6, f = i & 63;
            size_t g = ((size_t)(b0 + row) * Ssz + s) * Fsz + f;
            float xl = X[g], ml = Mm[g];
            float zh = sZhat[i], xh = sXhat[i], be = sBeta[i];
            float d2 = xl - zh;
            num_loc += d2 * d2 * ml;
            float ch = be * zh + (1.0f - be) * xh;
            float d3 = xl - ch;
            num_loc += d3 * d3 * ml;
            float cc = ml * xl + (1.0f - ml) * ch;
            out[g] = cc;
            sAcat[row * 128 + f]      = (__bf16)cc;
            sAcat[row * 128 + 64 + f] = (__bf16)ml;
        }
        // deterministic per-step loss reduction: wave shfl-reduce -> LDS -> thread 0
        {
            float rn = num_loc, rm = ms_loc;
#pragma unroll
            for (int off = 16; off > 0; off >>= 1) {
                rn += __shfl_xor(rn, off, 32);
                rm += __shfl_xor(rm, off, 32);
            }
            if (lane == 0) { sRed[wv * 2] = rn; sRed[wv * 2 + 1] = rm; }
        }
        __syncthreads();
        if (tid == 0) {
            float n = 0.0f, ms = 0.0f;
#pragma unroll
            for (int w = 0; w < 8; ++w) { n += sRed[w * 2]; ms += sRed[w * 2 + 1]; }
            wsNum[s * NBLK + blk] = n;
            wsMs [s * NBLK + blk] = ms;
        }

        // (g) LSTM gates = [cc,m] @ Wih^T + h @ Whh^T + (bih+bhh).  8 waves x 4 N-tiles.
        {
            v16bf aC[4], aH[4];
#pragma unroll
            for (int ki = 0; ki < 4; ++ki) {
                aC[ki] = lds_frag(sAcat, 128, ln, ki * 32, hi);
                aH[ki] = lds_frag(sHbf,  128, ln, ki * 32, hi);
            }
#pragma unroll
            for (int j = 0; j < 4; ++j) {
                int n = (wv * 4 + j) * 16 + ln;
                v8f acc = {0.f,0.f,0.f,0.f,0.f,0.f,0.f,0.f};
#pragma unroll
                for (int ki = 0; ki < 4; ++ki) acc = wmma_bf16(aC[ki], lds_frag(sWih, 128, n, ki * 32, hi), acc);
#pragma unroll
                for (int ki = 0; ki < 4; ++ki) acc = wmma_bf16(aH[ki], lds_frag(sWhh, 128, n, ki * 32, hi), acc);
                float bb = sBiasG[n];
#pragma unroll
                for (int r = 0; r < 8; ++r) sGates[(r + hi * 8) * 512 + n] = (__bf16)(acc[r] + bb);
            }
        }
        __syncthreads();

        // (h) LSTM state update
        for (int i = tid; i < ROWS * Hsz; i += NTHREADS) {
            int row = i >> 7, hc = i & 127;
            const __bf16* gr = sGates + row * 512;
            float gi = (float)gr[hc], gf = (float)gr[128 + hc];
            float gg = (float)gr[256 + hc], go = (float)gr[384 + hc];
            float c = sigmf(gf) * sCf[i] + sigmf(gi) * tanhf(gg);
            sCf[i] = c;
            float hv = sigmf(go) * tanhf(c);
            sHf[i] = hv;
            sHbf[i] = (__bf16)hv;
        }
        __syncthreads();
    }
}

__global__ void __launch_bounds__(Ssz)
rits_loss_kernel(const float* __restrict__ wsNum, const float* __restrict__ wsMs, float* __restrict__ lossOut)
{
    __shared__ float red[Ssz];
    int s = threadIdx.x;
    float n = 0.0f, ms = 0.0f;
#pragma unroll
    for (int b = 0; b < NBLK; ++b) { n += wsNum[s * NBLK + b]; ms += wsMs[s * NBLK + b]; }
    red[s] = n / (ms + 1e-5f);
    __syncthreads();
    for (int off = Ssz / 2; off > 0; off >>= 1) {
        if (s < off) red[s] += red[s + off];
        __syncthreads();
    }
    if (s == 0) *lossOut = red[0] / (float)Ssz;
}

extern "C" void kernel_launch(void* const* d_in, const int* in_sizes, int n_in,
                              void* d_out, int out_size, void* d_ws, size_t ws_size,
                              hipStream_t stream) {
    const float* X   = (const float*)d_in[0];
    const float* Mm  = (const float*)d_in[1];
    const float* T   = (const float*)d_in[2];
    const float* Wdh = (const float*)d_in[3];
    const float* bdh = (const float*)d_in[4];
    const float* Wdm = (const float*)d_in[5];
    const float* bdm = (const float*)d_in[6];
    const float* Wtr = (const float*)d_in[7];
    const float* btr = (const float*)d_in[8];
    const float* Wfr = (const float*)d_in[9];
    const float* bfr = (const float*)d_in[10];
    const float* Wwc = (const float*)d_in[11];
    const float* bwc = (const float*)d_in[12];
    const float* Wih = (const float*)d_in[13];
    const float* Whh = (const float*)d_in[14];
    const float* bih = (const float*)d_in[15];
    const float* bhh = (const float*)d_in[16];

    float* out   = (float*)d_out;
    float* wsNum = (float*)d_ws;                 // [S][NBLK]
    float* wsMs  = wsNum + (size_t)Ssz * NBLK;   // [S][NBLK]

    (void)hipFuncSetAttribute((const void*)rits_scan_kernel,
                              hipFuncAttributeMaxDynamicSharedMemorySize, LDS_BYTES);

    rits_scan_kernel<<<NBLK, NTHREADS, LDS_BYTES, stream>>>(
        X, Mm, T, Wdh, bdh, Wdm, bdm, Wtr, btr, Wfr, bfr, Wwc, bwc,
        Wih, Whh, bih, bhh, out, wsNum, wsMs);

    rits_loss_kernel<<<1, Ssz, 0, stream>>>(wsNum, wsMs, out + (size_t)Bsz * Ssz * Fsz);
}
